// NemotronHMambaDecoderLayer_10746008175009
// MI455X (gfx1250) — matrix operand order repaired and checked by reading, not compiled, not run on hardware
//
#include <hip/hip_runtime.h>
#include <hip/hip_bf16.h>

// ---------------- constants ----------------
#define LSEQ   512
#define HDIM   2048
#define NH     64
#define HD     64
#define INTER  4096
#define GG     4
#define NN     64
#define KCONV  4
#define CONVD  4608              // INTER + 2*G*N
#define PROJD  8768              // INTER + CONV + NH
#define DTOFF  8704              // INTER + CONV

typedef __attribute__((ext_vector_type(16))) __bf16        v16bf;
typedef __attribute__((ext_vector_type(8)))  float         v8f;
typedef __attribute__((ext_vector_type(8)))  unsigned int  v8u;

// round-to-nearest-even f32 -> bf16 (as u16)
__device__ __forceinline__ unsigned int f2bf(float f) {
    unsigned int u = __float_as_uint(f);
    return (u + 0x7FFFu + ((u >> 16) & 1u)) >> 16;
}

__device__ __forceinline__ v16bf asbf(v8u u) { return __builtin_bit_cast(v16bf, u); }

// ============================================================
// K1: x = hidden + residual ; new_residual = x ; hs = rmsnorm(x)*w
//     write hs as packed bf16 WMMA A-fragments (Kdim=2048, KB=64)
//     A-frag layout (16-bit A 16x32): lane = m%16 + 16*((kk>>3)&1),
//     dword j = ((kk>>4)<<2) + ((kk&7)>>1), K-even in [15:0].
// ============================================================
__global__ void k1_addnorm(const float* __restrict__ hidden,
                           const float* __restrict__ residual,
                           const float* __restrict__ norm_w,
                           float* __restrict__ newres,
                           unsigned int* __restrict__ Afrag)
{
    const int m   = blockIdx.x;       // 0..511
    const int tid = threadIdx.x;      // 0..255
    const int k0  = tid * 8;

    const float* hp = hidden   + (long)m * HDIM;
    const float* rp = residual + (long)m * HDIM;
    float*       nr = newres   + (long)m * HDIM;

    float v[8];
    float ss = 0.f;
    #pragma unroll
    for (int i = 0; i < 8; ++i) {
        float x = hp[k0 + i] + rp[k0 + i];
        nr[k0 + i] = x;
        v[i] = x;
        ss += x * x;
    }

    __shared__ float red[256];
    red[tid] = ss; __syncthreads();
    #pragma unroll
    for (int s = 128; s > 0; s >>= 1) {
        if (tid < s) red[tid] += red[tid + s];
        __syncthreads();
    }
    const float rs = rsqrtf(red[0] / (float)HDIM + 1e-5f);

    unsigned int d[4];
    #pragma unroll
    for (int i = 0; i < 4; ++i) {
        float a = v[2*i]   * rs * norm_w[k0 + 2*i];
        float b = v[2*i+1] * rs * norm_w[k0 + 2*i+1];
        d[i] = f2bf(a) | (f2bf(b) << 16);
    }
    const int mt = m >> 4, mr = m & 15;
    const int kb = k0 >> 5, kk = k0 & 31;
    const int lane = mr + (((kk >> 3) & 1) << 4);
    const int j0   = (kk >> 4) << 2;
    unsigned int* p = Afrag + (long)(mt * 64 + kb) * 256 + lane * 8 + j0;
    *(uint4*)p = make_uint4(d[0], d[1], d[2], d[3]);
}

// ============================================================
// K2: convert fp32 weight (Kdim x Ndim, row-major) into packed
//     bf16 WMMA B-fragments. B-frag (16-bit B 32x16): lane =
//     (n&15) + 16*((kk>>3)&1), dword j = ((kk>>4)<<2) + ((kk&7)>>1), K-even low half.
// ============================================================
__global__ void k_pack_w(const float* __restrict__ W,
                         unsigned int* __restrict__ Bfrag,
                         int Kdim, int Ndim)
{
    const long t = blockIdx.x * (long)blockDim.x + threadIdx.x;
    const long total = ((long)Kdim >> 1) * Ndim;
    if (t >= total) return;
    const int n  = (int)(t % Ndim);
    const int k0 = (int)(t / Ndim) * 2;

    const float f0 = W[(long)k0 * Ndim + n];
    const float f1 = W[(long)(k0 + 1) * Ndim + n];
    const unsigned int d = f2bf(f0) | (f2bf(f1) << 16);

    const int kb = k0 >> 5, kk = k0 & 31;
    const int lane = (n & 15) + (((kk >> 3) & 1) << 4);
    const int j    = ((kk >> 4) << 2) + ((kk & 7) >> 1);
    const int KB   = Kdim >> 5;
    Bfrag[(long)((n >> 4) * KB + kb) * 256 + lane * 8 + j] = d;
}

// ============================================================
// K3/K8: bf16 WMMA GEMM on packed fragments.
// One wave: 16 rows x 64 cols, KB iterations of 4 wmma ops.
// Software-pipelined: iteration k+1's 10x b128 loads are issued
// before iteration k's WMMAs so s_wait_loadcnt becomes partial.
// ============================================================
__global__ void k_gemm_bf16(const unsigned int* __restrict__ Afrag,
                            const unsigned int* __restrict__ Bfrag,
                            float* __restrict__ C,
                            int Mtiles, int KB, int Ndim, int Ngroups)
{
    const int lane = threadIdx.x & 31;
    const int wave = blockIdx.x * (blockDim.x >> 5) + (threadIdx.x >> 5);
    const int mt = wave % Mtiles;
    const int ng = wave / Mtiles;
    if (ng >= Ngroups) return;   // wave-uniform

    v8f acc0 = {}, acc1 = {}, acc2 = {}, acc3 = {};

    const unsigned int* ap = Afrag + (long)(mt * KB) * 256 + lane * 8;
    const unsigned int* b0 = Bfrag + (long)((ng * 4 + 0) * KB) * 256 + lane * 8;
    const unsigned int* b1 = Bfrag + (long)((ng * 4 + 1) * KB) * 256 + lane * 8;
    const unsigned int* b2 = Bfrag + (long)((ng * 4 + 2) * KB) * 256 + lane * 8;
    const unsigned int* b3 = Bfrag + (long)((ng * 4 + 3) * KB) * 256 + lane * 8;

    // prologue: fragments for kb = 0
    v8u a_c  = *(const v8u*)ap;
    v8u b0_c = *(const v8u*)b0;
    v8u b1_c = *(const v8u*)b1;
    v8u b2_c = *(const v8u*)b2;
    v8u b3_c = *(const v8u*)b3;

    #pragma unroll 2
    for (int kb = 0; kb < KB - 1; ++kb) {
        ap += 256; b0 += 256; b1 += 256; b2 += 256; b3 += 256;
        // issue next-iteration loads first (overlap with WMMA below)
        v8u a_n  = *(const v8u*)ap;
        v8u b0_n = *(const v8u*)b0;
        v8u b1_n = *(const v8u*)b1;
        v8u b2_n = *(const v8u*)b2;
        v8u b3_n = *(const v8u*)b3;

        acc0 = __builtin_amdgcn_wmma_f32_16x16x32_bf16(false, asbf(a_c), false, asbf(b0_c), (short)0, acc0, false, false);
        acc1 = __builtin_amdgcn_wmma_f32_16x16x32_bf16(false, asbf(a_c), false, asbf(b1_c), (short)0, acc1, false, false);
        acc2 = __builtin_amdgcn_wmma_f32_16x16x32_bf16(false, asbf(a_c), false, asbf(b2_c), (short)0, acc2, false, false);
        acc3 = __builtin_amdgcn_wmma_f32_16x16x32_bf16(false, asbf(a_c), false, asbf(b3_c), (short)0, acc3, false, false);

        a_c = a_n; b0_c = b0_n; b1_c = b1_n; b2_c = b2_n; b3_c = b3_n;
    }
    // epilogue iteration
    acc0 = __builtin_amdgcn_wmma_f32_16x16x32_bf16(false, asbf(a_c), false, asbf(b0_c), (short)0, acc0, false, false);
    acc1 = __builtin_amdgcn_wmma_f32_16x16x32_bf16(false, asbf(a_c), false, asbf(b1_c), (short)0, acc1, false, false);
    acc2 = __builtin_amdgcn_wmma_f32_16x16x32_bf16(false, asbf(a_c), false, asbf(b2_c), (short)0, acc2, false, false);
    acc3 = __builtin_amdgcn_wmma_f32_16x16x32_bf16(false, asbf(a_c), false, asbf(b3_c), (short)0, acc3, false, false);

    // C/D layout: lane 0-15 -> M=r, lane 16-31 -> M=r+8; N = lane&15
    const int row0 = mt * 16 + ((lane >> 4) << 3);
    const int col  = ng * 64 + (lane & 15);
    #pragma unroll
    for (int r = 0; r < 8; ++r) {
        float* cp = C + (long)(row0 + r) * Ndim + col;
        cp[0]  = acc0[r];
        cp[16] = acc1[r];
        cp[32] = acc2[r];
        cp[48] = acc3[r];
    }
}

// ============================================================
// K4: depthwise causal conv(K=4) + bias + SiLU over xBC columns
// ============================================================
__global__ void k4_conv(const float* __restrict__ proj,
                        const float* __restrict__ conv_w,
                        const float* __restrict__ conv_b,
                        float* __restrict__ xh,
                        float* __restrict__ BC)
{
    const long t = blockIdx.x * (long)blockDim.x + threadIdx.x;
    if (t >= (long)LSEQ * CONVD) return;
    const int c = (int)(t % CONVD);
    const int l = (int)(t / CONVD);

    float acc = conv_b[c];
    #pragma unroll
    for (int k = 0; k < KCONV; ++k) {
        const int ls = l + k - (KCONV - 1);
        const float x = (ls >= 0) ? proj[(long)ls * PROJD + INTER + c] : 0.f;
        acc += x * conv_w[c * KCONV + k];
    }
    const float s = acc / (1.f + __expf(-acc));
    if (c < INTER) xh[(long)l * INTER + c] = s;
    else           BC[(long)l * 512 + (c - INTER)] = s;   // [B(256) | C(256)]
}

// ============================================================
// K4b: dt = softplus(dt_raw + bias); dA = exp(dt * -exp(A_log))
// ============================================================
__global__ void k4b_dt(const float* __restrict__ proj,
                       const float* __restrict__ dt_bias,
                       const float* __restrict__ A_log,
                       float* __restrict__ dtp,
                       float* __restrict__ dAv)
{
    const int t = blockIdx.x * blockDim.x + threadIdx.x;
    if (t >= LSEQ * NH) return;
    const int h = t & 63;
    const int l = t >> 6;
    const float x  = proj[(long)l * PROJD + DTOFF + h] + dt_bias[h];
    const float sp = (x > 20.f) ? x : log1pf(__expf(x));
    const float A  = -__expf(A_log[h]);
    dtp[t] = sp;
    dAv[t] = __expf(sp * A);
}

// ============================================================
// K5: sequential selective scan. block = head (64), thread = hd dim
//     (64). Each thread keeps 64-wide state in registers; B/C
//     broadcast through LDS per step.
// ============================================================
__global__ void k5_scan(const float* __restrict__ xh,
                        const float* __restrict__ BC,
                        const float* __restrict__ dtp,
                        const float* __restrict__ dAv,
                        const float* __restrict__ Dp,
                        float* __restrict__ y)
{
    const int h = blockIdx.x;     // 0..63
    const int d = threadIdx.x;    // 0..63
    const int g = h >> 4;         // NH/G = 16 heads per group

    float hstate[NN];
    #pragma unroll
    for (int n = 0; n < NN; ++n) hstate[n] = 0.f;

    __shared__ float sB[NN];
    __shared__ float sC[NN];
    const float Dh = Dp[h];

    for (int l = 0; l < LSEQ; ++l) {
        sB[d] = BC[(long)l * 512 + g * NN + d];
        sC[d] = BC[(long)l * 512 + 256 + g * NN + d];
        __syncthreads();

        const float dA_t = dAv[l * NH + h];
        const float xv   = xh[(long)l * INTER + h * HD + d];
        const float dtx  = dtp[l * NH + h] * xv;

        float acc = 0.f;
        #pragma unroll
        for (int n = 0; n < NN; ++n) {
            hstate[n] = hstate[n] * dA_t + dtx * sB[n];
            acc += hstate[n] * sC[n];
        }
        y[(long)l * INTER + h * HD + d] = acc + xv * Dh;
        __syncthreads();
    }
}

// ============================================================
// K6: y *= silu(z); rmsnorm(gn_w); pack bf16 A-frags (Kdim=4096)
// ============================================================
__global__ void k6_gate_norm(const float* __restrict__ y,
                             const float* __restrict__ proj,
                             const float* __restrict__ gn_w,
                             unsigned int* __restrict__ Afrag)
{
    const int m   = blockIdx.x;
    const int tid = threadIdx.x;
    const int i0  = tid * 16;

    const float* yp = y    + (long)m * INTER;
    const float* zp = proj + (long)m * PROJD;   // z = first INTER cols

    float v[16];
    float ss = 0.f;
    #pragma unroll
    for (int i = 0; i < 16; ++i) {
        const float z = zp[i0 + i];
        const float val = yp[i0 + i] * (z / (1.f + __expf(-z)));
        v[i] = val;
        ss += val * val;
    }

    __shared__ float red[256];
    red[tid] = ss; __syncthreads();
    #pragma unroll
    for (int s = 128; s > 0; s >>= 1) {
        if (tid < s) red[tid] += red[tid + s];
        __syncthreads();
    }
    const float rs = rsqrtf(red[0] / (float)INTER + 1e-5f);

    const int mt = m >> 4, mr = m & 15;
    #pragma unroll
    for (int gidx = 0; gidx < 2; ++gidx) {
        const int k0 = i0 + gidx * 8;
        unsigned int dw[4];
        #pragma unroll
        for (int i = 0; i < 4; ++i) {
            float a = v[gidx*8 + 2*i]   * rs * gn_w[k0 + 2*i];
            float b = v[gidx*8 + 2*i+1] * rs * gn_w[k0 + 2*i+1];
            dw[i] = f2bf(a) | (f2bf(b) << 16);
        }
        const int kb = k0 >> 5, kk = k0 & 31;
        const int lane = mr + (((kk >> 3) & 1) << 4);
        const int j0   = (kk >> 4) << 2;
        unsigned int* p = Afrag + (long)(mt * 128 + kb) * 256 + lane * 8 + j0;
        *(uint4*)p = make_uint4(dw[0], dw[1], dw[2], dw[3]);
    }
}

// ============================================================
// launcher
// ============================================================
extern "C" void kernel_launch(void* const* d_in, const int* in_sizes, int n_in,
                              void* d_out, int out_size, void* d_ws, size_t ws_size,
                              hipStream_t stream)
{
    const float* hidden     = (const float*)d_in[0];
    const float* residual   = (const float*)d_in[1];
    const float* norm_w     = (const float*)d_in[2];
    const float* in_proj_w  = (const float*)d_in[3];
    const float* conv_w     = (const float*)d_in[4];
    const float* conv_b     = (const float*)d_in[5];
    const float* A_log      = (const float*)d_in[6];
    const float* dt_bias    = (const float*)d_in[7];
    const float* Dp         = (const float*)d_in[8];
    const float* gn_w       = (const float*)d_in[9];
    const float* out_proj_w = (const float*)d_in[10];

    float* out    = (float*)d_out;                     // [512,2048]
    float* newres = out + (long)LSEQ * HDIM;           // [512,2048]

    char* ws = (char*)d_ws;
    size_t o = 0;
    auto carve = [&](size_t bytes) { char* p = ws + o; o += (bytes + 255) & ~(size_t)255; return p; };

    unsigned int* hsA  = (unsigned int*)carve((size_t)LSEQ * HDIM * 2);          // 2 MB bf16 A-frags
    unsigned int* Wb   = (unsigned int*)carve((size_t)HDIM * PROJD * 2);         // 36 MB (reused for W2)
    float*        proj = (float*)carve((size_t)LSEQ * PROJD * 4);                // 18 MB
    float*        xh   = (float*)carve((size_t)LSEQ * INTER * 4);                // 8.4 MB
    float*        BC   = (float*)carve((size_t)LSEQ * 512 * 4);                  // 1 MB
    float*        dtp  = (float*)carve((size_t)LSEQ * NH * 4);
    float*        dAv  = (float*)carve((size_t)LSEQ * NH * 4);
    float*        ybuf = (float*)carve((size_t)LSEQ * INTER * 4);                // 8.4 MB
    unsigned int* yA   = (unsigned int*)carve((size_t)LSEQ * INTER * 2);         // 4.2 MB

    // 1. add + rmsnorm -> new_residual + packed hs
    k1_addnorm<<<LSEQ, 256, 0, stream>>>(hidden, residual, norm_w, newres, hsA);

    // 2. pack in_proj_w (2048 x 8768) -> bf16 B-frags
    k_pack_w<<<(int)(((long)HDIM / 2 * PROJD) / 256), 256, 0, stream>>>(in_proj_w, Wb, HDIM, PROJD);

    // 3. GEMM1: 512x2048 @ 2048x8768 ; Mtiles=32, KB=64, Ngroups=137
    k_gemm_bf16<<<548, 256, 0, stream>>>(hsA, Wb, proj, 32, 64, PROJD, PROJD / 64);

    // 4. conv + silu, dt/dA
    k4_conv<<<(int)(((long)LSEQ * CONVD) / 256), 256, 0, stream>>>(proj, conv_w, conv_b, xh, BC);
    k4b_dt<<<(LSEQ * NH) / 256, 256, 0, stream>>>(proj, dt_bias, A_log, dtp, dAv);

    // 5. selective scan
    k5_scan<<<NH, HD, 0, stream>>>(xh, BC, dtp, dAv, Dp, ybuf);

    // 6. gate + groupnorm -> packed y
    k6_gate_norm<<<LSEQ, 256, 0, stream>>>(ybuf, proj, gn_w, yA);

    // 7. pack out_proj_w (4096 x 2048) -> bf16 B-frags (reuse Wb)
    k_pack_w<<<(int)(((long)INTER / 2 * HDIM) / 256), 256, 0, stream>>>(out_proj_w, Wb, INTER, HDIM);

    // 8. GEMM2: 512x4096 @ 4096x2048 ; Mtiles=32, KB=128, Ngroups=32
    k_gemm_bf16<<<128, 256, 0, stream>>>(yA, Wb, out, 32, 128, HDIM, HDIM / 64);
}
